// TransformerLayer_78116865179728
// MI455X (gfx1250) — matrix operand order repaired and checked by reading, not compiled, not run on hardware
//
#include <hip/hip_runtime.h>
#include <math.h>

typedef _Float16 v16h __attribute__((ext_vector_type(16)));
typedef float    v8f  __attribute__((ext_vector_type(8)));
typedef unsigned int u32x4 __attribute__((ext_vector_type(4)));
typedef int          i32x4 __attribute__((ext_vector_type(4)));
typedef int          i32x8 __attribute__((ext_vector_type(8)));

#define BS   2
#define NN   256
#define DD   320
#define DEE  64
#define FFXN 256
#define ROWS (BS * NN)   // 512
#define ETPITCH 68       // 64 payload + 4 pad dwords per tile row in LDS

// Tensor Data Mover path (6-arg builtin form, per compile probe).
#if defined(__gfx1250__) && __has_builtin(__builtin_amdgcn_tensor_load_to_lds) && \
    __has_builtin(__builtin_amdgcn_s_wait_tensorcnt)
#define USE_TDM 1
#else
#define USE_TDM 0
#endif

// ---------------------------------------------------------------------------
// Generic tiled WMMA GEMM:  C[M,N] = A[M,K] @ B[K,N] + bias, compile-time
// row-mask / ReLU epilogue.  f32 in/out, f16 WMMA compute.
// Workgroup = 128 threads (4 waves); 16(M) x 64(N) tile; K steps of 32.
// B is staged transposed so each lane's fragment is 2 contiguous b128 reads.
// ---------------------------------------------------------------------------
template <bool RELU, bool MASKED>
__global__ __launch_bounds__(128)
void wmma_gemm(const float* __restrict__ A, const float* __restrict__ B,
               const float* __restrict__ bias, float* __restrict__ C,
               int M, int K, int N, const float* __restrict__ rowmask)
{
    __shared__ _Float16 Ast[16][40];    // 16x32 used, rows padded to 40
    __shared__ _Float16 BstT[64][40];   // [n][k] transposed, 32 K used

    const int tid  = threadIdx.x;
    const int wave = tid >> 5;
    const int lane = tid & 31;
    const int hi   = (lane >= 16) ? 1 : 0;
    const int lm   = lane & 15;

    const int m0  = blockIdx.x * 16;
    const int nb0 = blockIdx.y * 64;
    const int n0  = nb0 + wave * 16;

    v8f acc = {};

    for (int k0 = 0; k0 < K; k0 += 32) {
        __syncthreads();
        for (int e = tid; e < 16 * 32; e += 128) {          // A tile 16x32
            int r = e >> 5, c = e & 31;
            Ast[r][c] = (_Float16)A[(m0 + r) * K + (k0 + c)];
        }
        for (int e = tid; e < 32 * 64; e += 128) {          // B tile 32x64 (transposed)
            int kk = e >> 6, n = e & 63;
            BstT[n][kk] = (_Float16)B[(k0 + kk) * N + nb0 + n];
        }
        __syncthreads();

        v16h af, bf;
#pragma unroll
        for (int t = 0; t < 16; t++)                        // 16 contiguous halves
            af[t] = Ast[lm][t + (hi ? 16 : 0)];
#pragma unroll
        for (int t = 0; t < 16; t++)                        // 16 contiguous halves
            bf[t] = BstT[wave * 16 + lm][t + (hi ? 16 : 0)];

        acc = __builtin_amdgcn_wmma_f32_16x16x32_f16(
            false, af, false, bf, (short)0, acc, false, false);
    }

#pragma unroll
    for (int r = 0; r < 8; r++) {
        int row = m0 + r + (hi ? 8 : 0);
        int col = n0 + lm;
        float v = acc[r] + bias[col];
        if (RELU)   v = v > 0.f ? v : 0.f;
        if (MASKED) v *= rowmask[row];
        C[row * N + col] = v;
    }
}

// ---------------------------------------------------------------------------
// TDM issue: load a 16x64 f32 tile (row stride 64 elems) into LDS with
// 4-dword padding after each 64-dword row (LDS row pitch = 68 dwords).
// D# layout per CDNA5 ISA; 6-arg builtin form.
// ---------------------------------------------------------------------------
__device__ __forceinline__ void tdm_load_tile(const float* gsrc, unsigned lds_byte_off)
{
#if USE_TDM
    unsigned long long ga = (unsigned long long)(uintptr_t)gsrc;
    u32x4 g0 = {
        1u,                                                  // count=1 (valid user D#)
        lds_byte_off,                                        // lds_addr
        (unsigned)ga,                                        // global_addr[31:0]
        ((unsigned)(ga >> 32) & 0x01FFFFFFu) | (2u << 30)    // global_addr[56:32] | type=2
    };
    i32x8 g1 = {
        (int)((2u << 16) | (1u << 20) | (5u << 22) | (3u << 25)),
        //    data_size=4B | pad_en | pad_interval=64dw | pad_amount=4dw
        (int)(64u << 16),          // tensor_dim0 = 64 (bits 63:48)
        (int)(256u << 16),         // tensor_dim0 hi=0 | tensor_dim1 = 256
        (int)(64u << 16),          // tensor_dim1 hi=0 | tile_dim0 = 64
        16,                        // tile_dim1 = 16 | tile_dim2 = 0
        64,                        // tensor_dim0_stride = 64
        0, 0                       // tensor_dim1_stride = 0 (2D)
    };
    i32x4 gz4 = {0, 0, 0, 0};
    i32x8 gz8 = {0, 0, 0, 0, 0, 0, 0, 0};
    __builtin_amdgcn_tensor_load_to_lds(g0, g1, gz4, gz4, gz8, 0);
#else
    (void)gsrc; (void)lds_byte_off;
#endif
}

// ---------------------------------------------------------------------------
// Fused edge-projection + per-channel flash softmax + weighted-V.
// One workgroup (320 threads = 10 waves) per (b, i) query row.
// E tiles arrive via double-buffered TDM (overlapped with the consume phase);
// 40 WMMA 16x16 jobs per tile spread over 10 waves, weight fragments in regs.
// ---------------------------------------------------------------------------
__global__ __launch_bounds__(320)
void attn_kernel(const float* __restrict__ E, const float* __restrict__ mask,
                 const float* __restrict__ Wem, const float* __restrict__ bem,
                 const float* __restrict__ Wea, const float* __restrict__ bea,
                 const float* __restrict__ Qb, const float* __restrict__ Kb,
                 const float* __restrict__ Vb, float* __restrict__ WV)
{
    __shared__ float EtF[2][16][ETPITCH];   // double-buffered f32 E tiles (~8.5 KB)
    __shared__ float Y1[16][DD];            // 20 KB  (E @ Wem tile)
    __shared__ float Y2[16][DD];            // 20 KB  (E @ Wea tile)

    const int tid  = threadIdx.x;
    const int wave = tid >> 5;
    const int lane = tid & 31;
    const int hi   = (lane >= 16) ? 1 : 0;
    const int lm   = lane & 15;

    const int b = blockIdx.x / NN;
    const int i = blockIdx.x % NN;
    const float scale = 0.15811388300841898f; // 1/sqrt(DF=40)

    // Preload this wave's 4 weight B-fragments (loop invariant).
    v16h bfr[4][2];
#pragma unroll
    for (int j = 0; j < 4; j++) {
        int jb = wave * 4 + j;
        const float* Wsel = (jb < 20) ? Wem : Wea;
        int n0 = 16 * (jb % 20);
#pragma unroll
        for (int ks = 0; ks < 2; ks++)
#pragma unroll
            for (int t = 0; t < 16; t++)
                bfr[j][ks][t] = (_Float16)Wsel[(ks * 32 + t + (hi ? 16 : 0)) * DD + n0 + lm];
    }

    const int   c   = tid;                       // channel 0..319
    const float xmi = mask[b * NN + i];
    const float qv  = Qb[(b * NN + i) * DD + c];
    const float bm  = bem[c];
    const float ba  = bea[c];

    float m = -INFINITY, s = 0.f, acc = 0.f;
    const float* Ebase = E + (size_t)(b * NN + i) * NN * DEE;
    const int NT = NN / 16;

#if USE_TDM
    if (wave == 0)
        tdm_load_tile(Ebase, (unsigned)(uintptr_t)&EtF[0][0][0]);
#endif

    for (int t = 0; t < NT; t++) {
        const int cur = t & 1;
#if USE_TDM
        if (wave == 0 && t + 1 < NT)
            tdm_load_tile(Ebase + (size_t)(t + 1) * 16 * DEE,
                          (unsigned)(uintptr_t)&EtF[1 ^ cur][0][0]);
        if (t + 1 < NT)                       // uniform branch; constant wait args
            __builtin_amdgcn_s_wait_tensorcnt(1);
        else
            __builtin_amdgcn_s_wait_tensorcnt(0);
#else
        for (int e = tid; e < 16 * DEE; e += 320) {
            int jj = e >> 6, kk = e & 63;
            EtF[cur][jj][kk] = Ebase[(t * 16 + jj) * DEE + kk];
        }
#endif
        __syncthreads();

        // 40 WMMA jobs: 16 keys x 16 channels, K = 64 (2 steps of 32)
#pragma unroll
        for (int j = 0; j < 4; j++) {
            int jb = wave * 4 + j;
            int n0 = 16 * (jb % 20);
            v8f cacc = {};
#pragma unroll
            for (int ks = 0; ks < 2; ks++) {
                v16h af;
#pragma unroll
                for (int tt = 0; tt < 16; tt++)
                    af[tt] = (_Float16)EtF[cur][lm]
                              [ks * 32 + tt + (hi ? 16 : 0)];
                cacc = __builtin_amdgcn_wmma_f32_16x16x32_f16(
                    false, af, false, bfr[j][ks], (short)0, cacc, false, false);
            }
            float (*Yx)[DD] = (jb < 20) ? Y1 : Y2;
#pragma unroll
            for (int r = 0; r < 8; r++)
                Yx[r + (hi ? 8 : 0)][n0 + lm] = cacc[r];
        }
        __syncthreads();

        // online softmax over this tile's 16 keys (one channel per thread);
        // overlaps with the TDM fill of the other E buffer.
        for (int jj = 0; jj < 16; jj++) {
            int jg = t * 16 + jj;
            float xmj = mask[b * NN + jg];
            float em  = xmi * xmj;
            float e1  = (Y1[jj][c] + bm) * em;
            float e2  = (Y2[jj][c] + ba) * em;
            float kv  = Kb[(b * NN + jg) * DD + c];
            float y   = qv * kv * scale * (e1 + 1.f) + e2;
            if (xmj <= 0.f) y = -1.0e9f;
            if (y > m) { float corr = __expf(m - y); s *= corr; acc *= corr; m = y; }
            float p = __expf(y - m);
            s   += p;
            acc += p * Vb[(b * NN + jg) * DD + c];
        }
        __syncthreads();
    }
    WV[(b * NN + i) * DD + c] = acc / s;
}

// ---------------------------------------------------------------------------
// Fused residual add + LayerNorm over last dim (DD=320). One row per block.
// ---------------------------------------------------------------------------
__global__ __launch_bounds__(320)
void add_ln(const float* __restrict__ A, const float* __restrict__ Bv,
            const float* __restrict__ g, const float* __restrict__ beta,
            float* __restrict__ out)
{
    __shared__ float red[DD];
    const int row = blockIdx.x, t = threadIdx.x;
    float v = A[row * DD + t] + Bv[row * DD + t];

    red[t] = v;
    __syncthreads();
    for (int off = 256; off > 0; off >>= 1) {
        if (t < off && t + off < DD) red[t] += red[t + off];
        __syncthreads();
    }
    float mu = red[0] * (1.f / DD);
    __syncthreads();

    float d = v - mu;
    red[t] = d * d;
    __syncthreads();
    for (int off = 256; off > 0; off >>= 1) {
        if (t < off && t + off < DD) red[t] += red[t + off];
        __syncthreads();
    }
    float var = red[0] * (1.f / DD);
    out[row * DD + t] = d * rsqrtf(var + 1e-5f) * g[t] + beta[t];
}

// ---------------------------------------------------------------------------
extern "C" void kernel_launch(void* const* d_in, const int* in_sizes, int n_in,
                              void* d_out, int out_size, void* d_ws, size_t ws_size,
                              hipStream_t stream)
{
    const float* X    = (const float*)d_in[0];
    const float* E    = (const float*)d_in[1];
    const float* mask = (const float*)d_in[2];
    const float* Wq   = (const float*)d_in[3];  const float* bq  = (const float*)d_in[4];
    const float* Wk   = (const float*)d_in[5];  const float* bk  = (const float*)d_in[6];
    const float* Wv   = (const float*)d_in[7];  const float* bv  = (const float*)d_in[8];
    const float* Wem  = (const float*)d_in[9];  const float* bem = (const float*)d_in[10];
    const float* Wea  = (const float*)d_in[11]; const float* bea = (const float*)d_in[12];
    const float* Wo   = (const float*)d_in[13]; const float* bo  = (const float*)d_in[14];
    const float* W1   = (const float*)d_in[15]; const float* b1  = (const float*)d_in[16];
    const float* W2   = (const float*)d_in[17]; const float* b2  = (const float*)d_in[18];
    const float* g1   = (const float*)d_in[19]; const float* be1 = (const float*)d_in[20];
    const float* g2   = (const float*)d_in[21]; const float* be2 = (const float*)d_in[22];
    float* out = (float*)d_out;

    float* ws  = (float*)d_ws;
    float* Qb  = ws; ws += ROWS * DD;
    float* Kb  = ws; ws += ROWS * DD;
    float* Vb  = ws; ws += ROWS * DD;
    float* WVb = ws; ws += ROWS * DD;
    float* NXb = ws; ws += ROWS * DD;
    float* X1b = ws; ws += ROWS * DD;
    float* Hb  = ws; ws += ROWS * FFXN;
    float* F2b = ws; ws += ROWS * DD;

    dim3 blk(128);
    dim3 gD(ROWS / 16, DD / 64);     // 32 x 5
    dim3 gF(ROWS / 16, FFXN / 64);   // 32 x 4

    // Q/K/V projections (masked)
    wmma_gemm<false, true><<<gD, blk, 0, stream>>>(X, Wq, bq, Qb, ROWS, DD, DD, mask);
    wmma_gemm<false, true><<<gD, blk, 0, stream>>>(X, Wk, bk, Kb, ROWS, DD, DD, mask);
    wmma_gemm<false, true><<<gD, blk, 0, stream>>>(X, Wv, bv, Vb, ROWS, DD, DD, mask);

    // fused edge projection + per-channel softmax + weighted V
    attn_kernel<<<ROWS, 320, 0, stream>>>(E, mask, Wem, bem, Wea, bea,
                                          Qb, Kb, Vb, WVb);

    // output projection (masked) + residual LN1
    wmma_gemm<false, true><<<gD, blk, 0, stream>>>(WVb, Wo, bo, NXb, ROWS, DD, DD, mask);
    add_ln<<<ROWS, 320, 0, stream>>>(X, NXb, g1, be1, X1b);

    // feed-forward + residual LN2
    wmma_gemm<true, false><<<gF, blk, 0, stream>>>(X1b, W1, b1, Hb, ROWS, DD, FFXN, nullptr);
    wmma_gemm<false, false><<<gD, blk, 0, stream>>>(Hb, W2, b2, F2b, ROWS, FFXN, DD, nullptr);
    add_ln<<<ROWS, 320, 0, stream>>>(X1b, F2b, g2, be2, out);
}
